// TPA_36885179138722
// MI455X (gfx1250) — matrix operand order repaired and checked by reading, not compiled, not run on hardware
//
#include <hip/hip_runtime.h>

// ---------------------------------------------------------------------------
// TPA forward for MI455X (gfx1250, wave32, WMMA, async-to-LDS copies).
//   1. f32->bf16 convert of x; f32->bf16 transposed convert of weights
//   2. WMMA projection GEMMs (bf16 in, f32 acc), BK=64, double-buffered LDS
//      tiles filled with GLOBAL_LOAD_ASYNC_TO_LDS_B128
//   3. RoPE + rank contraction -> q,k,v (bf16); v transposed to [BH][HD][S]
//   4. Flash attention on v_wmma_f32_16x16x32_bf16, base-2 online softmax,
//      double-buffered async K/V staging
// ---------------------------------------------------------------------------

typedef __attribute__((ext_vector_type(16))) __bf16 v16bf;
typedef __attribute__((ext_vector_type(8)))  float  v8f;

#define Hh   32
#define Ss   2048
#define Dd   2048
#define HDd  64
#define QRr  12
#define RRr  2
#define Bb   2
#define MTOK (Bb * Ss)          // 4096 tokens
#define LOG2E 1.4426950408889634f

struct U2x16B { uint4 a, b; };  // 32B payload for one bf16 fragment

__device__ __forceinline__ unsigned short f32_to_bf16(float f) {
  union { float f; unsigned u; } v; v.f = f;
  unsigned u = v.u;
  u += 0x7fffu + ((u >> 16) & 1u);      // round-to-nearest-even
  return (unsigned short)(u >> 16);
}

// CDNA5 async global->LDS copy (ASYNCcnt tracked), 16B per active lane.
__device__ __forceinline__ void async_b128(unsigned lds_off, const void* gptr) {
  asm volatile("global_load_async_to_lds_b128 %0, %1, off"
               :: "v"(lds_off), "v"((unsigned long long)(uintptr_t)gptr)
               : "memory");
}
__device__ __forceinline__ void wait_async0() {
  asm volatile("s_wait_asynccnt 0x0" ::: "memory");
}
__device__ __forceinline__ unsigned lds_off32(const void* p) {
  return (unsigned)(uintptr_t)p;        // AS(3) ptrtoint -> byte offset
}

// A-frag (16x32, row-major [m][k] source): halves 0-7 -> K kb..kb+7 (+8 hi
// lanes), halves 8-15 -> K kb+16..kb+23 (+8 hi lanes).
__device__ __forceinline__ v16bf
load_a_frag(const unsigned short* lds, int row0, int pad, int kbase, int lane) {
  int m  = row0 + (lane & 15);
  int kh = kbase + (lane >> 4) * 8;
  U2x16B p;
  p.a = *(const uint4*)&lds[m * pad + kh];
  p.b = *(const uint4*)&lds[m * pad + kh + 16];
  return __builtin_bit_cast(v16bf, p);
}

// B-frag (32x16, from transposed [n][k] source): halves 0-7 -> K kb..kb+7
// (+8 hi lanes), halves 8-15 -> K kb+16..kb+23 (+8 hi lanes).
__device__ __forceinline__ v16bf
load_b_frag(const unsigned short* lds, int col0, int pad, int kbase, int lane) {
  int n  = col0 + (lane & 15);
  int kb = kbase + (lane >> 4) * 8;
  U2x16B p;
  p.a = *(const uint4*)&lds[n * pad + kb];
  p.b = *(const uint4*)&lds[n * pad + kb + 16];
  return __builtin_bit_cast(v16bf, p);
}

#define WMMA_BF16(a, b, c) \
  __builtin_amdgcn_wmma_f32_16x16x32_bf16(false, (a), false, (b), (short)0, (c), false, false)

#define V8F_ZERO (v8f){0.f,0.f,0.f,0.f,0.f,0.f,0.f,0.f}

// ------------------------------- convert -----------------------------------
__global__ void __launch_bounds__(256)
convert_f32_bf16(const float* __restrict__ src, unsigned short* __restrict__ dst,
                 int n4) {
  int i = blockIdx.x * 256 + threadIdx.x;
  if (i >= n4) return;
  float4 f = ((const float4*)src)[i];
  uint2 p;
  p.x = (unsigned)f32_to_bf16(f.x) | ((unsigned)f32_to_bf16(f.y) << 16);
  p.y = (unsigned)f32_to_bf16(f.z) | ((unsigned)f32_to_bf16(f.w) << 16);
  ((uint2*)dst)[i] = p;
}

// Convert + transpose weight: src f32 [K=2048][N] -> dst bf16 [N][2048].
__global__ void __launch_bounds__(256)
transpose_conv_w(const float* __restrict__ src, unsigned short* __restrict__ dst,
                 int N) {
  __shared__ unsigned short t[32][33];
  const int n0 = blockIdx.x * 32, k0 = blockIdx.y * 32;
  const int tx = threadIdx.x, ty = threadIdx.y;     // 32 x 8
#pragma unroll
  for (int i = 0; i < 4; ++i)
    t[ty + i * 8][tx] = f32_to_bf16(src[(size_t)(k0 + ty + i * 8) * N + n0 + tx]);
  __syncthreads();
#pragma unroll
  for (int i = 0; i < 4; ++i)
    dst[(size_t)(n0 + ty + i * 8) * Dd + k0 + tx] = t[tx][ty + i * 8];
}

// Transpose v: src bf16 [BH][S][64] -> dst bf16 [BH][64][S].
__global__ void __launch_bounds__(256)
transpose_v(const unsigned short* __restrict__ src,
            unsigned short* __restrict__ dst) {
  __shared__ unsigned short t[32][33];
  const int bh = blockIdx.z;
  const int s0 = blockIdx.x * 32, d0 = blockIdx.y * 32;
  const int tx = threadIdx.x, ty = threadIdx.y;     // 32 x 8
#pragma unroll
  for (int i = 0; i < 4; ++i)
    t[ty + i * 8][tx] = src[((size_t)bh * Ss + s0 + ty + i * 8) * HDd + d0 + tx];
  __syncthreads();
#pragma unroll
  for (int i = 0; i < 4; ++i)
    dst[((size_t)bh * HDd + d0 + ty + i * 8) * Ss + s0 + tx] = t[tx][ty + i * 8];
}

// ------------------------------- GEMM --------------------------------------
// C[M=4096, N] = Xb[4096, 2048] @ Wtb^T, with Wtb bf16 [N][2048].
// Block 256 (8 waves), tile 128x64, BK=64, double-buffered async staging.
#define GPAD 72   // 144B row stride: 16B-aligned, conflict-free b128 frag reads

__global__ void __launch_bounds__(256)
gemm_bf16(const unsigned short* __restrict__ Xb,
          const unsigned short* __restrict__ Wtb,
          float* __restrict__ C, int N) {
  __shared__ unsigned short ldsX[2][128 * GPAD];   // [m][k]
  __shared__ unsigned short ldsW[2][64 * GPAD];    // [n][k]
  const int tid  = threadIdx.x;
  const int lane = tid & 31, wave = tid >> 5;
  const int mw = wave >> 1, nw = wave & 1;
  const int rowBase = blockIdx.y * 128;
  const int colBase = blockIdx.x * 64;

  const int sr = tid >> 3, sq = (tid & 7) * 8;     // staging coords
  const unsigned short* Xrow = Xb + (size_t)rowBase * Dd;
  const unsigned short* Wrow = Wtb + (size_t)colBase * Dd;

  auto issue = [&](int k0, int buf) {
#pragma unroll
    for (int it = 0; it < 4; ++it) {               // X: 128 rows x 8 uint4
      int r = sr + it * 32;
      async_b128(lds_off32(&ldsX[buf][r * GPAD + sq]),
                 Xrow + (size_t)r * Dd + k0 + sq);
    }
#pragma unroll
    for (int it = 0; it < 2; ++it) {               // W: 64 rows x 8 uint4
      int r = sr + it * 32;
      async_b128(lds_off32(&ldsW[buf][r * GPAD + sq]),
                 Wrow + (size_t)r * Dd + k0 + sq);
    }
  };

  v8f acc[2][2];
#pragma unroll
  for (int i = 0; i < 2; ++i)
#pragma unroll
    for (int j = 0; j < 2; ++j) acc[i][j] = V8F_ZERO;

  issue(0, 0);
  for (int t = 0; t < Dd / 64; ++t) {
    wait_async0();
    __syncthreads();
    if (t + 1 < Dd / 64) issue((t + 1) * 64, (t + 1) & 1);
    const unsigned short* lX = ldsX[t & 1];
    const unsigned short* lW = ldsW[t & 1];

    v16bf af[2][2], bf[2][2];                      // [kf][mt] / [kf][nt]
#pragma unroll
    for (int kf = 0; kf < 2; ++kf)
#pragma unroll
      for (int mt = 0; mt < 2; ++mt)
        af[kf][mt] = load_a_frag(lX, mw * 32 + mt * 16, GPAD, kf * 32, lane);
#pragma unroll
    for (int kf = 0; kf < 2; ++kf)
#pragma unroll
      for (int nt = 0; nt < 2; ++nt)
        bf[kf][nt] = load_b_frag(lW, nw * 32 + nt * 16, GPAD, kf * 32, lane);

    // both K-chunks chained per accumulator: D->C accumulate, no reg shuffles
#pragma unroll
    for (int mt = 0; mt < 2; ++mt)
#pragma unroll
      for (int nt = 0; nt < 2; ++nt) {
        acc[mt][nt] = WMMA_BF16(af[0][mt], bf[0][nt], acc[mt][nt]);
        acc[mt][nt] = WMMA_BF16(af[1][mt], bf[1][nt], acc[mt][nt]);
      }
    __syncthreads();
  }

#pragma unroll
  for (int mt = 0; mt < 2; ++mt)
#pragma unroll
    for (int nt = 0; nt < 2; ++nt) {
      int row0 = rowBase + mw * 32 + mt * 16 + ((lane >> 4) << 3);
      int col  = colBase + nw * 32 + nt * 16 + (lane & 15);
#pragma unroll
      for (int r = 0; r < 8; ++r)
        C[(size_t)(row0 + r) * N + col] = acc[mt][nt][r];
    }
}

// -------------------------- RoPE + rank contraction -------------------------
__global__ void __launch_bounds__(64)
rope_contract(const float* __restrict__ Aq, const float* __restrict__ Ak,
              const float* __restrict__ Av, const float* __restrict__ Bq,
              const float* __restrict__ Bk, const float* __restrict__ Bv,
              const float* __restrict__ cosb, const float* __restrict__ sinb,
              unsigned short* __restrict__ qo, unsigned short* __restrict__ ko,
              unsigned short* __restrict__ vo) {
  const int tok = blockIdx.x;
  const int b = tok >> 11;
  const int s = tok & 2047;
  const int d = threadIdx.x;

  __shared__ float sA[512];         // Aq[384] | Ak[64] | Av[64]
#pragma unroll
  for (int it = 0; it < 6; ++it)
    sA[d + it * 64] = Aq[(size_t)tok * 384 + d + it * 64];
  sA[384 + d] = Ak[(size_t)tok * 64 + d];
  sA[448 + d] = Av[(size_t)tok * 64 + d];
  __syncthreads();

  const float c  = cosb[s * 32 + (d >> 1)];
  const float sn = sinb[s * 32 + (d >> 1)];
  const bool odd = d & 1;
  const int  de  = d & ~1;

  float bq[QRr], bk[RRr], bv[RRr];
  const float* Bqt = Bq + (size_t)tok * QRr * HDd;
  const float* Bkt = Bk + (size_t)tok * RRr * HDd;
  const float* Bvt = Bv + (size_t)tok * RRr * HDd;
#pragma unroll
  for (int r = 0; r < QRr; ++r) {
    float e = Bqt[r * 64 + de], o = Bqt[r * 64 + de + 1];
    bq[r] = odd ? (e * sn + o * c) : (e * c - o * sn);
  }
#pragma unroll
  for (int r = 0; r < RRr; ++r) {
    float e = Bkt[r * 64 + de], o = Bkt[r * 64 + de + 1];
    bk[r] = odd ? (e * sn + o * c) : (e * c - o * sn);
    bv[r] = Bvt[r * 64 + d];
  }

#pragma unroll 4
  for (int h = 0; h < Hh; ++h) {
    float aq = 0.f;
#pragma unroll
    for (int r = 0; r < QRr; ++r) aq += sA[h * QRr + r] * bq[r];
    float kk = (sA[384 + h * 2] * bk[0] + sA[384 + h * 2 + 1] * bk[1]) * 0.5f;
    float vv = (sA[448 + h * 2] * bv[0] + sA[448 + h * 2 + 1] * bv[1]) * 0.5f;
    size_t base = ((size_t)(b * Hh + h) * Ss + s) * HDd + d;
    qo[base] = f32_to_bf16(aq * (1.0f / QRr));
    ko[base] = f32_to_bf16(kk);
    vo[base] = f32_to_bf16(vv);
  }
}

// ------------------------------ Flash attention -----------------------------
// Grid: (S/128, B*H). Block 256 (8 waves); wave w owns query rows w*16..+15.
// Base-2 online softmax; causal mask as -1e9 (matches reference's mask add).
#define QPAD 72

__global__ void __launch_bounds__(256)
flash_attn(const unsigned short* __restrict__ Qb,
           const unsigned short* __restrict__ Kb,
           const unsigned short* __restrict__ Vt,   // [BH][64][S]
           float* __restrict__ Out) {
  __shared__ unsigned short ldsQ[128 * QPAD];
  __shared__ unsigned short ldsK[2][64 * QPAD];     // [key][dim]
  __shared__ unsigned short ldsVT[2][64 * QPAD];    // [dim][key]
  __shared__ unsigned short ldsP[128 * QPAD];       // wave-private P rows

  const int tid = threadIdx.x, lane = tid & 31, wave = tid >> 5;
  const int qt = blockIdx.x;
  const int bh = blockIdx.y;
  const size_t headBase = (size_t)bh * Ss * HDd;

  const unsigned short* Qh  = Qb + headBase + (size_t)qt * 128 * HDd;
  const unsigned short* KhB = Kb + headBase;
  const unsigned short* VtB = Vt + headBase;        // [64][S]

  const int sr = tid >> 3, sq = (tid & 7) * 8;

  auto issueKV = [&](int j, int buf) {
#pragma unroll
    for (int it = 0; it < 2; ++it) {                // K: 64 rows x 8 uint4
      int r = sr + it * 32;
      async_b128(lds_off32(&ldsK[buf][r * QPAD + sq]),
                 KhB + ((size_t)j * 64 + r) * HDd + sq);
    }
#pragma unroll
    for (int it = 0; it < 2; ++it) {                // VT: 64 dims x 8 uint4
      int r = sr + it * 32;
      async_b128(lds_off32(&ldsVT[buf][r * QPAD + sq]),
                 VtB + (size_t)r * Ss + j * 64 + sq);
    }
  };

  // stage Q (async, once) + first K/V tile, then wait -> Q frags live in regs
#pragma unroll
  for (int it = 0; it < 4; ++it) {                  // Q: 128 rows x 8 uint4
    int r = sr + it * 32;
    async_b128(lds_off32(&ldsQ[r * QPAD + sq]), Qh + (size_t)r * HDd + sq);
  }
  issueKV(0, 0);
  wait_async0();
  __syncthreads();

  const v16bf qa0 = load_a_frag(ldsQ, wave * 16, QPAD, 0, lane);
  const v16bf qa1 = load_a_frag(ldsQ, wave * 16, QPAD, 32, lane);

  v8f o[4];
  float m_i[8], l_i[8];
#pragma unroll
  for (int nt = 0; nt < 4; ++nt) o[nt] = V8F_ZERO;
#pragma unroll
  for (int r = 0; r < 8; ++r) { m_i[r] = -3.0e38f; l_i[r] = 0.0f; }

  const int rowg0 = qt * 128 + wave * 16 + ((lane >> 4) << 3);
  const int jEnd = 2 * qt + 1;
  const float SC = 0.125f * LOG2E;                  // 1/sqrt(HD), base-2 domain

  for (int j = 0; j <= jEnd; ++j) {
    if (j + 1 <= jEnd) issueKV(j + 1, (j + 1) & 1);
    const unsigned short* lK = ldsK[j & 1];
    const unsigned short* lV = ldsVT[j & 1];

    // ---- S = Q K^T (8 WMMAs per wave, chained per accumulator) ----
    v8f sf[4];
#pragma unroll
    for (int nt = 0; nt < 4; ++nt) {
      v16bf b0 = load_b_frag(lK, nt * 16, QPAD, 0, lane);
      v16bf b1 = load_b_frag(lK, nt * 16, QPAD, 32, lane);
      sf[nt] = WMMA_BF16(qa0, b0, V8F_ZERO);
      sf[nt] = WMMA_BF16(qa1, b1, sf[nt]);
    }

    // ---- scale (base-2) + causal mask ----
    const bool needMask = (j * 64 + 63) > (qt * 128);
#pragma unroll
    for (int nt = 0; nt < 4; ++nt) {
      int colg = j * 64 + nt * 16 + (lane & 15);
#pragma unroll
      for (int r = 0; r < 8; ++r) {
        float sv = sf[nt][r] * SC;
        if (needMask && colg > rowg0 + r) sv -= 1.0e9f;
        sf[nt][r] = sv;
      }
    }

    // ---- base-2 online softmax (rows live on 16-lane halves) ----
    float rmax[8];
#pragma unroll
    for (int r = 0; r < 8; ++r)
      rmax[r] = fmaxf(fmaxf(sf[0][r], sf[1][r]), fmaxf(sf[2][r], sf[3][r]));
#pragma unroll
    for (int off = 1; off < 16; off <<= 1)
#pragma unroll
      for (int r = 0; r < 8; ++r)
        rmax[r] = fmaxf(rmax[r], __shfl_xor(rmax[r], off, 32));

    float scl[8];
#pragma unroll
    for (int r = 0; r < 8; ++r) {
      float mn = fmaxf(m_i[r], rmax[r]);
      scl[r] = exp2f(m_i[r] - mn);
      m_i[r] = mn;
    }
#pragma unroll
    for (int nt = 0; nt < 4; ++nt)
#pragma unroll
      for (int r = 0; r < 8; ++r)
        sf[nt][r] = exp2f(sf[nt][r] - m_i[r]);

    float rsum[8];
#pragma unroll
    for (int r = 0; r < 8; ++r)
      rsum[r] = sf[0][r] + sf[1][r] + sf[2][r] + sf[3][r];
#pragma unroll
    for (int off = 1; off < 16; off <<= 1)
#pragma unroll
      for (int r = 0; r < 8; ++r)
        rsum[r] += __shfl_xor(rsum[r], off, 32);
#pragma unroll
    for (int r = 0; r < 8; ++r) l_i[r] = l_i[r] * scl[r] + rsum[r];
#pragma unroll
    for (int nt = 0; nt < 4; ++nt)
#pragma unroll
      for (int r = 0; r < 8; ++r) o[nt][r] *= scl[r];

    // ---- P -> LDS (wave-private rows; same-wave DScnt ordering) ----
#pragma unroll
    for (int nt = 0; nt < 4; ++nt)
#pragma unroll
      for (int r = 0; r < 8; ++r)
        ldsP[(wave * 16 + r + ((lane >> 4) << 3)) * QPAD + nt * 16 + (lane & 15)]
            = f32_to_bf16(sf[nt][r]);

    // ---- O += P V (8 WMMAs per wave, chained per accumulator) ----
    {
      v16bf pa0 = load_a_frag(ldsP, wave * 16, QPAD, 0, lane);
      v16bf pa1 = load_a_frag(ldsP, wave * 16, QPAD, 32, lane);
#pragma unroll
      for (int nt = 0; nt < 4; ++nt) {
        v16bf b0 = load_b_frag(lV, nt * 16, QPAD, 0, lane);
        v16bf b1 = load_b_frag(lV, nt * 16, QPAD, 32, lane);
        o[nt] = WMMA_BF16(pa0, b0, o[nt]);
        o[nt] = WMMA_BF16(pa1, b1, o[nt]);
      }
    }

    if (j < jEnd) {            // next tile arrived; all waves done with buf
      wait_async0();
      __syncthreads();
    }
  }

  // ---- epilogue: O / l -> out [B][S][H*HD] ----
  const int b = bh >> 5, h = bh & 31;
#pragma unroll
  for (int nt = 0; nt < 4; ++nt) {
    int col = h * 64 + nt * 16 + (lane & 15);
#pragma unroll
    for (int r = 0; r < 8; ++r) {
      int row = qt * 128 + wave * 16 + r + ((lane >> 4) << 3);
      Out[(size_t)(b * Ss + row) * (Hh * HDd) + col] = o[nt][r] / l_i[r];
    }
  }
}

// ------------------------------- launcher ----------------------------------
extern "C" void kernel_launch(void* const* d_in, const int* in_sizes, int n_in,
                              void* d_out, int out_size, void* d_ws, size_t ws_size,
                              hipStream_t stream) {
  const float* x   = (const float*)d_in[0];
  const float* fc  = (const float*)d_in[2];
  const float* fs  = (const float*)d_in[3];
  const float* wAq = (const float*)d_in[5];
  const float* wAk = (const float*)d_in[6];
  const float* wAv = (const float*)d_in[7];
  const float* wBq = (const float*)d_in[8];
  const float* wBk = (const float*)d_in[9];
  const float* wBv = (const float*)d_in[10];
  float* out = (float*)d_out;

  char* ws = (char*)d_ws;
  size_t off = 0;
  auto alloc = [&](size_t bytes) -> char* {
    char* p = ws + off; off += (bytes + 255) & ~(size_t)255; return p;
  };
  unsigned short* xb   = (unsigned short*)alloc((size_t)MTOK * Dd * 2);
  unsigned short* wAqb = (unsigned short*)alloc((size_t)Dd * 384 * 2);  // [N][K]
  unsigned short* wAkb = (unsigned short*)alloc((size_t)Dd * 64 * 2);
  unsigned short* wAvb = (unsigned short*)alloc((size_t)Dd * 64 * 2);
  unsigned short* wBqb = (unsigned short*)alloc((size_t)Dd * 768 * 2);
  unsigned short* wBkb = (unsigned short*)alloc((size_t)Dd * 128 * 2);
  unsigned short* wBvb = (unsigned short*)alloc((size_t)Dd * 128 * 2);
  float* pAq = (float*)alloc((size_t)MTOK * 384 * 4);
  float* pAk = (float*)alloc((size_t)MTOK * 64 * 4);
  float* pAv = (float*)alloc((size_t)MTOK * 64 * 4);
  float* pBq = (float*)alloc((size_t)MTOK * 768 * 4);
  float* pBk = (float*)alloc((size_t)MTOK * 128 * 4);
  float* pBv = (float*)alloc((size_t)MTOK * 128 * 4);
  unsigned short* qb   = (unsigned short*)alloc((size_t)MTOK * Hh * HDd * 2);
  unsigned short* kb   = (unsigned short*)alloc((size_t)MTOK * Hh * HDd * 2);
  unsigned short* vtmp = (unsigned short*)alloc((size_t)MTOK * Hh * HDd * 2);
  unsigned short* vT   = (unsigned short*)alloc((size_t)MTOK * Hh * HDd * 2);
  // total 110 MiB of workspace

  convert_f32_bf16<<<dim3((MTOK * Dd / 4 + 255) / 256), dim3(256), 0, stream>>>(
      x, xb, MTOK * Dd / 4);

  auto tw = [&](const float* src, unsigned short* dst, int N) {
    transpose_conv_w<<<dim3(N / 32, Dd / 32), dim3(32, 8), 0, stream>>>(src, dst, N);
  };
  tw(wAq, wAqb, 384);
  tw(wAk, wAkb, 64);
  tw(wAv, wAvb, 64);
  tw(wBq, wBqb, 768);
  tw(wBk, wBkb, 128);
  tw(wBv, wBvb, 128);

  auto gemm = [&](const unsigned short* w, float* c, int N) {
    gemm_bf16<<<dim3(N / 64, MTOK / 128), dim3(256), 0, stream>>>(xb, w, c, N);
  };
  gemm(wAqb, pAq, 384);
  gemm(wAkb, pAk, 64);
  gemm(wAvb, pAv, 64);
  gemm(wBqb, pBq, 768);
  gemm(wBkb, pBk, 128);
  gemm(wBvb, pBv, 128);

  rope_contract<<<dim3(MTOK), dim3(64), 0, stream>>>(
      pAq, pAk, pAv, pBq, pBk, pBv, fc, fs, qb, kb, vtmp);

  transpose_v<<<dim3(Ss / 32, HDd / 32, Bb * Hh), dim3(32, 8), 0, stream>>>(
      vtmp, vT);

  flash_attn<<<dim3(Ss / 128, Bb * Hh), dim3(256), 0, stream>>>(qb, kb, vT, out);

  (void)in_sizes; (void)n_in; (void)out_size; (void)ws_size;
}
